// Perceiver_283467842555
// MI455X (gfx1250) — compile-verified, matrix operand order checked by reading, not run on hardware
//
#include <hip/hip_runtime.h>
#include <hip/hip_bf16.h>
#include <math.h>

#define B_  64
#define N_  128
#define C_  256
#define L_  512
#define D_  512
#define DEPTH_ 4
#define SPC_ 2

// GEMM tiling
#define BM 128
#define BN 128
#define BK 32
#define LDST 36   // padded LDS row stride in floats (multiple of 4 -> 16B alignment; spreads banks)

typedef __attribute__((ext_vector_type(16))) _Float16 v16h;
typedef __attribute__((ext_vector_type(8)))  float    v8f;

// ---------------- reductions (wave32) ----------------
__device__ __forceinline__ float warp_sum(float v) {
#pragma unroll
  for (int o = 16; o > 0; o >>= 1) v += __shfl_xor(v, o, 32);
  return v;
}
__device__ __forceinline__ float warp_max(float v) {
#pragma unroll
  for (int o = 16; o > 0; o >>= 1) v = fmaxf(v, __shfl_xor(v, o, 32));
  return v;
}
__device__ float block_sum(float v, float* sm) {
  const int lane = threadIdx.x & 31, wid = threadIdx.x >> 5, nw = (int)blockDim.x >> 5;
  __syncthreads();
  v = warp_sum(v);
  if (lane == 0) sm[wid] = v;
  __syncthreads();
  float r = (threadIdx.x < (unsigned)nw) ? sm[threadIdx.x] : 0.0f;
  if (wid == 0) { r = warp_sum(r); if (lane == 0) sm[0] = r; }
  __syncthreads();
  return sm[0];
}
__device__ float block_max(float v, float* sm) {
  const int lane = threadIdx.x & 31, wid = threadIdx.x >> 5, nw = (int)blockDim.x >> 5;
  __syncthreads();
  v = warp_max(v);
  if (lane == 0) sm[wid] = v;
  __syncthreads();
  float r = (threadIdx.x < (unsigned)nw) ? sm[threadIdx.x] : -3.4e38f;
  if (wid == 0) { r = warp_max(r); if (lane == 0) sm[0] = r; }
  __syncthreads();
  return sm[0];
}

// ------------- CDNA5 async global->LDS copy (16B per lane), ASYNCcnt-tracked -------------
__device__ __forceinline__ void async_g2l_b128(unsigned lds_off, const float* gaddr) {
  asm volatile("global_load_async_to_lds_b128 %0, %1, off"
               :: "v"(lds_off), "v"((unsigned long long)(uintptr_t)gaddr)
               : "memory");
}
__device__ __forceinline__ void wait_async0() { asm volatile("s_wait_asynccnt 0x0" ::: "memory"); }
__device__ __forceinline__ void wait_async4() { asm volatile("s_wait_asynccnt 0x4" ::: "memory"); }
__device__ __forceinline__ void wait_async8() { asm volatile("s_wait_asynccnt 0x8" ::: "memory"); }

// ---------------- WMMA GEMM ----------------
// D[m,n] = alpha * sum_k A[m,k]*B[k,n] (+bias[n]) (+Res[m,n]); transB=1: B stored [N,K].
// Batched over (b,h) via gridDim.z. Block = 8 waves, tile 128x128; wave tile 64x32 (4x2 WMMA).
// Double-buffered LDS; async copies of tile k+1 overlap WMMA on tile k.
__global__ __launch_bounds__(256)
void wmma_gemm_f16(const float* __restrict__ A, const float* __restrict__ Bm,
                   const float* __restrict__ bias, const float* __restrict__ Res,
                   float* __restrict__ C,
                   int M, int N, int K, int lda, int ldb, int ldc,
                   long Ab, long Ah, long Bb, long Bh, long Cb, long Ch,
                   int nh, int transB, float alpha)
{
  __shared__ float lsA[2][BM * LDST];   // A tiles [128 rows][32 k], padded, double-buffered
  __shared__ float lsB[2][BN * LDST];   // B tiles transposed [128 cols][32 k]

  const int z = blockIdx.z;
  const int b = z / nh, h = z - b * nh;
  A  += (long)b * Ab + (long)h * Ah;
  Bm += (long)b * Bb + (long)h * Bh;
  const long coff = (long)b * Cb + (long)h * Ch;

  const int bm0 = blockIdx.y * BM;
  const int bn0 = blockIdx.x * BN;

  const int lane = threadIdx.x;            // 0..31
  const int wav  = threadIdx.y;            // 0..7
  const int tid  = lane + (wav << 5);      // 0..255
  const int wm   = wav & 1;                // 2 waves along M (64 rows each)
  const int wn   = wav >> 1;               // 4 waves along N (32 cols each)
  const int r16  = lane & 15;
  const int hi   = lane >> 4;
  const int koff = hi << 3;

  v8f acc[4][2];
#pragma unroll
  for (int sm = 0; sm < 4; ++sm)
#pragma unroll
    for (int sn = 0; sn < 2; ++sn)
      acc[sm][sn] = (v8f){0.f,0.f,0.f,0.f,0.f,0.f,0.f,0.f};

  // staging map: 256 threads x 4 iters x 16B cover a 128x32 f32 tile
  const int kg  = tid & 7;                 // k-group of 4 floats
  const int rr  = tid >> 3;                // 0..31 (row, stepped by 32)
  const int n4  = (tid & 31) * 4;          // manual-B: n group of 4
  const int kk0 = tid >> 5;                // manual-B: k, stepped by 8
  const unsigned ldsA0 = (unsigned)(uintptr_t)&lsA[0][0];
  const unsigned ldsB0 = (unsigned)(uintptr_t)&lsB[0][0];
  const unsigned bufBytes = (unsigned)(BM * LDST) * 4u;

  float4 bq[4];  // manual-B pipeline registers

  // ---- staging helpers ----
  auto stageA = [&](int kt, int buf) {
#pragma unroll
    for (int it = 0; it < 4; ++it) {
      const int r = rr + 32 * it;
      if (bm0 + r < M)
        async_g2l_b128(ldsA0 + (unsigned)buf * bufBytes + (unsigned)(r * LDST + kg * 4) * 4u,
                       A + (long)(bm0 + r) * lda + kt * BK + kg * 4);
    }
  };
  auto stageBasync = [&](int kt, int buf) {  // transB: rows are n, K-contiguous
#pragma unroll
    for (int it = 0; it < 4; ++it) {
      const int n = rr + 32 * it;
      if (bn0 + n < N)
        async_g2l_b128(ldsB0 + (unsigned)buf * bufBytes + (unsigned)(n * LDST + kg * 4) * 4u,
                       Bm + (long)(bn0 + n) * ldb + kt * BK + kg * 4);
    }
  };
  auto loadBq = [&](int kt) {                // non-trans: coalesced float4 along n
#pragma unroll
    for (int it = 0; it < 4; ++it) {
      const int kk = kk0 + 8 * it;
      bq[it] = (bn0 + n4 < N)
                   ? *(const float4*)(Bm + (long)(kt * BK + kk) * ldb + bn0 + n4)
                   : make_float4(0.f, 0.f, 0.f, 0.f);
    }
  };
  auto storeBq = [&](int buf) {              // transpose-scatter into lsB[buf][n][k]
    float* d = &lsB[buf][0];
#pragma unroll
    for (int it = 0; it < 4; ++it) {
      const int kk = kk0 + 8 * it;
      d[(n4 + 0) * LDST + kk] = bq[it].x;
      d[(n4 + 1) * LDST + kk] = bq[it].y;
      d[(n4 + 2) * LDST + kk] = bq[it].z;
      d[(n4 + 3) * LDST + kk] = bq[it].w;
    }
  };

  const int nt = K / BK;   // K always a multiple of 32 here

  // ---- prologue: stage tile 0 into buffer 0 ----
  if (transB) {
    stageA(0, 0);
    stageBasync(0, 0);
  } else {
    stageA(0, 0);
    loadBq(0);
    storeBq(0);
  }

  for (int kt = 0; kt < nt; ++kt) {
    const int cur = kt & 1, nxt = cur ^ 1;
    const bool more = (kt + 1 < nt);

    // issue next tile's copies before waiting on the current one
    if (transB) {
      if (more) { stageA(kt + 1, nxt); stageBasync(kt + 1, nxt); wait_async8(); }
      else wait_async0();
    } else {
      if (more) { stageA(kt + 1, nxt); loadBq(kt + 1); wait_async4(); }
      else wait_async0();
    }
    __syncthreads();   // current tile visible to all waves

    // ---- all fragments converted up-front (independent register sets), then WMMA burst ----
    v16h bfr[2];
#pragma unroll
    for (int sn = 0; sn < 2; ++sn) {
      const float* pb = &lsB[cur][(wn * 32 + sn * 16 + r16) * LDST];
#pragma unroll
      for (int i = 0; i < 8; ++i) {
        bfr[sn][i]     = (_Float16)pb[koff + i];
        bfr[sn][i + 8] = (_Float16)pb[koff + 16 + i];
      }
    }
    v16h afr[4];
#pragma unroll
    for (int sm = 0; sm < 4; ++sm) {
      const float* pa = &lsA[cur][(wm * 64 + sm * 16 + r16) * LDST];
#pragma unroll
      for (int i = 0; i < 8; ++i) {
        afr[sm][i]     = (_Float16)pa[koff + i];
        afr[sm][i + 8] = (_Float16)pa[koff + 16 + i];
      }
    }
#pragma unroll
    for (int sm = 0; sm < 4; ++sm)
#pragma unroll
      for (int sn = 0; sn < 2; ++sn)
        acc[sm][sn] = __builtin_amdgcn_wmma_f32_16x16x32_f16(
            false, afr[sm], false, bfr[sn], (short)0, acc[sm][sn], false, false);

    if (more) {
      __syncthreads();               // all reads of buffers done before overwriting
      if (!transB) storeBq(nxt);     // LDS transpose of next tile (visible at next barrier)
    }
  }

  // ---- epilogue (tile guards are wave-uniform; WMMA already done) ----
  C += coff;
  const float* R = Res ? Res + coff : nullptr;
#pragma unroll
  for (int sm = 0; sm < 4; ++sm) {
    const int tm = bm0 + wm * 64 + sm * 16;
    if (tm >= M) continue;
    const int row0 = tm + (hi << 3);
#pragma unroll
    for (int sn = 0; sn < 2; ++sn) {
      const int tn = bn0 + wn * 32 + sn * 16;
      if (tn >= N) continue;
      const int col = tn + r16;
      const float bb = bias ? bias[col] : 0.0f;
#pragma unroll
      for (int r = 0; r < 8; ++r) {
        const long idx = (long)(row0 + r) * ldc + col;
        float v = acc[sm][sn][r] * alpha + bb;
        if (R) v += R[idx];
        C[idx] = v;
      }
    }
  }
}

// ---------------- elementwise kernels ----------------
__global__ __launch_bounds__(256)
void layernorm_rows(const float* __restrict__ x, const float* __restrict__ g,
                    const float* __restrict__ b, float* __restrict__ y, int dim)
{
  __shared__ float sm[8];
  const long row = blockIdx.x;
  const float* px = x + row * dim;
  float* py = y + row * dim;
  float s = 0.f;
  for (int i = threadIdx.x; i < dim; i += blockDim.x) s += px[i];
  const float mean = block_sum(s, sm) / (float)dim;
  float v = 0.f;
  for (int i = threadIdx.x; i < dim; i += blockDim.x) { float d = px[i] - mean; v += d * d; }
  const float rstd = rsqrtf(block_sum(v, sm) / (float)dim + 1e-5f);
  for (int i = threadIdx.x; i < dim; i += blockDim.x)
    py[i] = (px[i] - mean) * rstd * g[i] + b[i];
}

__global__ __launch_bounds__(256)
void softmax_rows(float* __restrict__ s, int W)
{
  __shared__ float sm[8];
  float* p = s + (long)blockIdx.x * W;
  float m = -3.4e38f;
  for (int i = threadIdx.x; i < W; i += blockDim.x) m = fmaxf(m, p[i]);
  m = block_max(m, sm);
  float acc = 0.f;
  for (int i = threadIdx.x; i < W; i += blockDim.x) { float e = expf(p[i] - m); p[i] = e; acc += e; }
  const float inv = 1.0f / block_sum(acc, sm);
  for (int i = threadIdx.x; i < W; i += blockDim.x) p[i] *= inv;
}

__global__ void bcast_latents(const float* __restrict__ lat, float* __restrict__ X,
                              long per, long total)
{
  long i = (long)blockIdx.x * blockDim.x + threadIdx.x;
  if (i < total) X[i] = lat[i % per];
}

// GEGLU: h=[rows,4096]; out[row,j] = h[row,j] * gelu_exact(h[row,2048+j]), j<2048
__global__ void geglu_act(const float* __restrict__ h, float* __restrict__ out, long rows)
{
  long i = (long)blockIdx.x * blockDim.x + threadIdx.x;
  const long total = rows * 2048;
  if (i >= total) return;
  const long row = i >> 11;
  const int  j   = (int)(i & 2047);
  const float a  = h[row * 4096 + j];
  const float gt = h[row * 4096 + 2048 + j];
  out[i] = a * (0.5f * gt * (1.0f + erff(gt * 0.70710678118654752f)));
}

__global__ void mean_pool(const float* __restrict__ X, float* __restrict__ P)
{
  const int i = blockIdx.x * blockDim.x + threadIdx.x;
  if (i >= B_ * D_) return;
  const int b = i / D_, d = i - b * D_;
  const float* px = X + (long)b * L_ * D_ + d;
  float s = 0.f;
  for (int n = 0; n < L_; ++n) s += px[(long)n * D_];
  P[i] = s * (1.0f / (float)L_);
}

__global__ __launch_bounds__(256)
void head_kernel(const float* __restrict__ P, const float* __restrict__ g,
                 const float* __restrict__ b, const float* __restrict__ w,
                 const float* __restrict__ hb, float* __restrict__ out)
{
  __shared__ float sm[8];
  const int bi = blockIdx.x;
  const float* px = P + (long)bi * D_;
  float s = 0.f;
  for (int i = threadIdx.x; i < D_; i += blockDim.x) s += px[i];
  const float mean = block_sum(s, sm) / (float)D_;
  float v = 0.f;
  for (int i = threadIdx.x; i < D_; i += blockDim.x) { float d = px[i] - mean; v += d * d; }
  const float rstd = rsqrtf(block_sum(v, sm) / (float)D_ + 1e-5f);
  float dot = 0.f;
  for (int i = threadIdx.x; i < D_; i += blockDim.x)
    dot += ((px[i] - mean) * rstd * g[i] + b[i]) * w[i];
  dot = block_sum(dot, sm);
  if (threadIdx.x == 0) out[bi] = dot + hb[0];
}

// ---------------- host-side launch helper ----------------
static inline void launch_gemm(hipStream_t s,
    const float* A, const float* Bm, const float* bias, const float* Res, float* C,
    int M, int N, int K, int lda, int ldb, int ldc,
    long Ab, long Ah, long Bb, long Bh, long Cb, long Ch,
    int nb, int nh, int transB, float alpha)
{
  dim3 grid((N + BN - 1) / BN, (M + BM - 1) / BM, nb * nh);
  dim3 blk(32, 8, 1);
  wmma_gemm_f16<<<grid, blk, 0, s>>>(A, Bm, bias, Res, C, M, N, K, lda, ldb, ldc,
                                     Ab, Ah, Bb, Bh, Cb, Ch, nh, transB, alpha);
}

extern "C" void kernel_launch(void* const* d_in, const int* in_sizes, int n_in,
                              void* d_out, int out_size, void* d_ws, size_t ws_size,
                              hipStream_t stream) {
  (void)in_sizes; (void)n_in; (void)out_size; (void)ws_size;
  const float* data      = (const float*)d_in[0];
  const float* latents   = (const float*)d_in[1];
  const float* ca_norm_g = (const float*)d_in[2];
  const float* ca_norm_b = (const float*)d_in[3];
  const float* ca_ctx_g  = (const float*)d_in[4];
  const float* ca_ctx_b  = (const float*)d_in[5];
  const float* ca_wq     = (const float*)d_in[6];
  const float* ca_wkv    = (const float*)d_in[7];
  const float* ca_wo     = (const float*)d_in[8];
  const float* ca_bo     = (const float*)d_in[9];
  const float* cf_norm_g = (const float*)d_in[10];
  const float* cf_norm_b = (const float*)d_in[11];
  const float* cf_w1     = (const float*)d_in[12];
  const float* cf_b1     = (const float*)d_in[13];
  const float* cf_w2     = (const float*)d_in[14];
  const float* cf_b2     = (const float*)d_in[15];
  const float* sa_norm_g = (const float*)d_in[16];
  const float* sa_norm_b = (const float*)d_in[17];
  const float* sa_wq     = (const float*)d_in[18];
  const float* sa_wkv    = (const float*)d_in[19];
  const float* sa_wo     = (const float*)d_in[20];
  const float* sa_bo     = (const float*)d_in[21];
  const float* sf_norm_g = (const float*)d_in[22];
  const float* sf_norm_b = (const float*)d_in[23];
  const float* sf_w1     = (const float*)d_in[24];
  const float* sf_b1     = (const float*)d_in[25];
  const float* sf_w2     = (const float*)d_in[26];
  const float* sf_b2     = (const float*)d_in[27];
  const float* head_g    = (const float*)d_in[28];
  const float* head_b    = (const float*)d_in[29];
  const float* head_w    = (const float*)d_in[30];
  const float* head_bias = (const float*)d_in[31];
  // d_in[32] = mask [B,N]: all-true in this configuration -> no-op in reference.

  float* out = (float*)d_out;

  // -------- workspace layout (floats) --------
  float* ws = (float*)d_ws;
  size_t off = 0;
  float* X   = ws + off; off += (size_t)B_ * L_ * D_;      //  64 MB latents state
  float* XN  = ws + off; off += (size_t)B_ * L_ * D_;      //  64 MB LN(x)
  float* CN  = ws + off; off += (size_t)B_ * N_ * C_;      //   8 MB LN(data)
  float* Q   = ws + off; off += (size_t)B_ * L_ * 512;     //  64 MB
  float* KV  = ws + off; off += (size_t)B_ * L_ * 1024;    // 128 MB
  float* O   = ws + off; off += (size_t)B_ * L_ * 512;     //  64 MB attn out
  float* BIG = ws + off; off += (size_t)B_ * 8 * L_ * L_;  // 537 MB: sim AND ffn hidden
  float* GA  = ws + off; off += (size_t)B_ * L_ * 2048;    // 268 MB gated
  float* PO  = ws + off; off += (size_t)B_ * D_;           // pooled

  const long LD = (long)L_ * D_;

  // x = broadcast(latents)
  {
    long total = (long)B_ * L_ * D_;
    bcast_latents<<<dim3((unsigned)((total + 255) / 256)), dim3(256), 0, stream>>>(
        latents, X, (long)L_ * D_, total);
  }

  for (int i = 0; i < DEPTH_; ++i) {
    // ---------------- cross attention ----------------
    layernorm_rows<<<dim3(B_ * L_), dim3(256), 0, stream>>>(X, ca_norm_g + (size_t)i * D_,
                                                            ca_norm_b + (size_t)i * D_, XN, D_);
    layernorm_rows<<<dim3(B_ * N_), dim3(256), 0, stream>>>(data, ca_ctx_g + (size_t)i * C_,
                                                            ca_ctx_b + (size_t)i * C_, CN, C_);
    // Q = XN @ wq   [B, L, 64]
    launch_gemm(stream, XN, ca_wq + (size_t)i * D_ * 64, nullptr, nullptr, Q,
                L_, 64, D_, D_, 64, 64, LD, 0, 0, 0, (long)L_ * 64, 0, B_, 1, 0, 1.0f);
    // KV = CN @ wkv [B, N, 128]  (k = cols 0..63, v = cols 64..127)
    launch_gemm(stream, CN, ca_wkv + (size_t)i * C_ * 128, nullptr, nullptr, KV,
                N_, 128, C_, C_, 128, 128, (long)N_ * C_, 0, 0, 0, (long)N_ * 128, 0, B_, 1, 0, 1.0f);
    // SIM = Q @ K^T * 1/sqrt(64)   [B, L, N]
    launch_gemm(stream, Q, KV, nullptr, nullptr, BIG,
                L_, N_, 64, 64, 128, N_, (long)L_ * 64, 0, (long)N_ * 128, 0,
                (long)L_ * N_, 0, B_, 1, 1, 0.125f);
    softmax_rows<<<dim3(B_ * L_), dim3(256), 0, stream>>>(BIG, N_);
    // O = SIM @ V   [B, L, 64]
    launch_gemm(stream, BIG, KV + 64, nullptr, nullptr, O,
                L_, 64, N_, N_, 128, 64, (long)L_ * N_, 0, (long)N_ * 128, 0,
                (long)L_ * 64, 0, B_, 1, 0, 1.0f);
    // X = O @ wo + bo + X
    launch_gemm(stream, O, ca_wo + (size_t)i * 64 * D_, ca_bo + (size_t)i * D_, X, X,
                L_, D_, 64, 64, D_, D_, (long)L_ * 64, 0, 0, 0, LD, 0, B_, 1, 0, 1.0f);

    // ---------------- cross GEGLU FF ----------------
    layernorm_rows<<<dim3(B_ * L_), dim3(256), 0, stream>>>(X, cf_norm_g + (size_t)i * D_,
                                                            cf_norm_b + (size_t)i * D_, XN, D_);
    launch_gemm(stream, XN, cf_w1 + (size_t)i * D_ * 4096, cf_b1 + (size_t)i * 4096,
                nullptr, BIG, L_, 4096, D_, D_, 4096, 4096, LD, 0, 0, 0,
                (long)L_ * 4096, 0, B_, 1, 0, 1.0f);
    {
      long total = (long)B_ * L_ * 2048;
      geglu_act<<<dim3((unsigned)((total + 255) / 256)), dim3(256), 0, stream>>>(
          BIG, GA, (long)B_ * L_);
    }
    launch_gemm(stream, GA, cf_w2 + (size_t)i * 2048 * D_, cf_b2 + (size_t)i * D_, X, X,
                L_, D_, 2048, 2048, D_, D_, (long)L_ * 2048, 0, 0, 0, LD, 0, B_, 1, 0, 1.0f);

    // ---------------- latent self attention blocks ----------------
    for (int j = 0; j < SPC_; ++j) {
      const size_t p = (size_t)i * SPC_ + j;
      layernorm_rows<<<dim3(B_ * L_), dim3(256), 0, stream>>>(X, sa_norm_g + p * D_,
                                                              sa_norm_b + p * D_, XN, D_);
      // Q = XN @ wq [B, L, 512] ; KV = XN @ wkv [B, L, 1024]
      launch_gemm(stream, XN, sa_wq + p * D_ * 512, nullptr, nullptr, Q,
                  L_, 512, D_, D_, 512, 512, LD, 0, 0, 0, (long)L_ * 512, 0, B_, 1, 0, 1.0f);
      launch_gemm(stream, XN, sa_wkv + p * D_ * 1024, nullptr, nullptr, KV,
                  L_, 1024, D_, D_, 1024, 1024, LD, 0, 0, 0, (long)L_ * 1024, 0, B_, 1, 0, 1.0f);
      // SIM[b,h] = Q[b,:,h*64:] @ K[b,:,h*64:]^T * 0.125     [B,8,L,L]
      launch_gemm(stream, Q, KV, nullptr, nullptr, BIG,
                  L_, L_, 64, 512, 1024, L_,
                  (long)L_ * 512, 64, (long)L_ * 1024, 64,
                  (long)8 * L_ * L_, (long)L_ * L_, B_, 8, 1, 0.125f);
      softmax_rows<<<dim3(B_ * 8 * L_), dim3(256), 0, stream>>>(BIG, L_);
      // O[b,:,h*64:] = SIM[b,h] @ V[b,:,512+h*64:]
      launch_gemm(stream, BIG, KV + 512, nullptr, nullptr, O,
                  L_, 64, L_, L_, 1024, 512,
                  (long)8 * L_ * L_, (long)L_ * L_, (long)L_ * 1024, 64,
                  (long)L_ * 512, 64, B_, 8, 0, 1.0f);
      // X = O @ wo + bo + X
      launch_gemm(stream, O, sa_wo + p * 512 * D_, sa_bo + p * D_, X, X,
                  L_, D_, 512, 512, D_, D_, (long)L_ * 512, 0, 0, 0, LD, 0, B_, 1, 0, 1.0f);

      // self GEGLU FF
      layernorm_rows<<<dim3(B_ * L_), dim3(256), 0, stream>>>(X, sf_norm_g + p * D_,
                                                              sf_norm_b + p * D_, XN, D_);
      launch_gemm(stream, XN, sf_w1 + p * D_ * 4096, sf_b1 + p * 4096, nullptr, BIG,
                  L_, 4096, D_, D_, 4096, 4096, LD, 0, 0, 0, (long)L_ * 4096, 0, B_, 1, 0, 1.0f);
      {
        long total = (long)B_ * L_ * 2048;
        geglu_act<<<dim3((unsigned)((total + 255) / 256)), dim3(256), 0, stream>>>(
            BIG, GA, (long)B_ * L_);
      }
      launch_gemm(stream, GA, sf_w2 + p * 2048 * D_, sf_b2 + p * D_, X, X,
                  L_, D_, 2048, 2048, D_, D_, (long)L_ * 2048, 0, 0, 0, LD, 0, B_, 1, 0, 1.0f);
    }
  }

  // pooled = mean over latents; out = LN(pooled) @ head_w + head_bias
  mean_pool<<<dim3((B_ * D_ + 255) / 256), dim3(256), 0, stream>>>(X, PO);
  head_kernel<<<dim3(B_), dim3(256), 0, stream>>>(PO, head_g, head_b, head_w, head_bias, out);
}
